// AdaptiveWindowDistillLoss_52063593562611
// MI455X (gfx1250) — matrix operand order repaired and checked by reading, not compiled, not run on hardware
//
#include <hip/hip_runtime.h>
#include <hip/hip_bf16.h>
#include <math.h>

#define H4    128
#define W4    512
#define CCH   96
#define DP1   64
#define V0R   85
#define NROW  43        // rows 85..127
#define BATCH 4
#define TPB   256

typedef __bf16 bf16_t;
typedef __bf16 v16bf __attribute__((ext_vector_type(16)));
typedef float  v8f   __attribute__((ext_vector_type(8)));
typedef unsigned int v4u __attribute__((ext_vector_type(4)));
typedef int    v8i   __attribute__((ext_vector_type(8)));
typedef int    v4i   __attribute__((ext_vector_type(4)));

// ---- LDS layout (dynamic shared) ----
// [0, 98304)              : bf16 FqR row  [512][96]
// [98304, 98304+196608)   : f32 TDM staging (FqR row raw) -- recycled after convert:
//    [98304, 98304+131072): f32 cv / prefix  [64][512]
//    [+131072, +2048)     : int  inval[512]
//    [+2048 .., 3072)     : float red[768] (seg sums / reduction)
#define SZ_FR      (W4 * CCH * 2)            // 98304
#define OFF_STAGE  SZ_FR
#define OFF_CVP    SZ_FR
#define SZ_CVP     (DP1 * W4 * 4)            // 131072
#define OFF_INVAL  (OFF_CVP + SZ_CVP)
#define OFF_RED    (OFF_INVAL + W4 * 4)
#define SMEM_BYTES (SZ_FR + W4 * CCH * 4)    // 294912 (288 KB, fits 320 KB WGP LDS)

#define ENT_CLIP_EXTRA 1.8420681e-7f         // -EPS*log(EPS), EPS=1e-8

#ifndef HAVE_TDM
#if defined(__has_builtin)
#if __has_builtin(__builtin_amdgcn_tensor_load_to_lds)
#define HAVE_TDM 1
#endif
#endif
#endif
#ifndef HAVE_TDM
#define HAVE_TDM 0
#endif

__global__ void awdl_init(float* ws) {
  ws[0] = 0.0f; ws[1] = 0.0f; ws[2] = 0.0f;
}

__global__ __launch_bounds__(TPB) void awdl_main(const float* __restrict__ FqL,
                                                 const float* __restrict__ FqR,
                                                 const float* __restrict__ Stud,
                                                 float* __restrict__ ws) {
  extern __shared__ char smem[];
  bf16_t* fr    = (bf16_t*)smem;
  float*  stage = (float*)(smem + OFF_STAGE);
  float*  cvp   = (float*)(smem + OFF_CVP);
  int*    inval = (int*)(smem + OFF_INVAL);
  float*  red   = (float*)(smem + OFF_RED);

  const int tid = threadIdx.x;
  const int bb  = blockIdx.x / NROW;
  const int hh  = V0R + (blockIdx.x % NROW);
  const size_t rowBase = ((size_t)bb * H4 + hh) * (size_t)W4;

  // ---------------- Phase 1: FqR row -> LDS (TDM async DMA), then f32->bf16 ----
#if HAVE_TDM
  if (tid < 32) {
    const unsigned long long ga = (unsigned long long)(uintptr_t)(FqR + rowBase * CCH);
    v4u g0;
    g0[0] = 1u;                               // count=1, user descriptor
    g0[1] = (unsigned)OFF_STAGE;              // lds_addr (bytes)
    g0[2] = (unsigned)(ga & 0xFFFFFFFFull);   // global_addr[31:0]
    g0[3] = (unsigned)((ga >> 32) & 0x1FFFFFFull) | (2u << 30);  // addr[56:32] | type=2
    v8i g1;
    g1[0] = 0x20000;                          // data_size=2 (4B), no multicast
    g1[1] = (int)0xC0000000;                  // tensor_dim0 = 49152 (low16 in bits 63:48)
    g1[2] = (1 << 16);                        // tensor_dim0 hi16=0 | tensor_dim1=1
    g1[3] = (int)0xC0000000;                  // tile_dim0 = 49152
    g1[4] = 0;                                // tile_dim1/2 unused (1D copy)
    g1[5] = 49152;                            // tensor_dim0_stride[31:0]
    g1[6] = 0; g1[7] = 0;
    v4i g2; g2[0]=0; g2[1]=0; g2[2]=0; g2[3]=0;
    v4i g3; g3[0]=0; g3[1]=0; g3[2]=0; g3[3]=0;
    v8i g4; g4[0]=0; g4[1]=0; g4[2]=0; g4[3]=0; g4[4]=0; g4[5]=0; g4[6]=0; g4[7]=0;
    __builtin_amdgcn_tensor_load_to_lds(g0, g1, g2, g3, g4, 0);
    __builtin_amdgcn_s_wait_tensorcnt(0);
  }
#else
  {
    const float* src = FqR + rowBase * CCH;
    for (int i = tid; i < W4 * CCH; i += TPB) stage[i] = src[i];
  }
#endif
  __syncthreads();
  for (int i = tid; i < W4 * CCH; i += TPB) fr[i] = (bf16_t)stage[i];
  __syncthreads();

  // ---------------- Phase 2: banded Gram matrix via bf16 WMMA ------------------
  // cv[d][w] = <FqL[w], FqR[w-d]>, 0<=d<64, w>=d
  {
    const int lane = tid & 31;
    const int wv   = tid >> 5;          // 8 waves
    const int Mrow = lane & 15;
    const int hi   = (lane >> 4) & 1;

    for (int t = 0; t < 4; ++t) {
      const int ti = wv + 8 * t;        // output w-tile
      // A tile (FqL rows 16*ti..+15, K=0..95) -> registers, contiguous k order k = 16*hi + s (same as B)
      const float* Ab = FqL + (rowBase + (size_t)(ti * 16 + Mrow)) * CCH;
      v16bf a0, a1, a2;
#pragma unroll
      for (int c = 0; c < 3; ++c) {
        v16bf av;
        const int klo = 32 * c + hi * 16;
        const int khi = 32 * c + hi * 16 + 8;
#pragma unroll
        for (int i = 0; i < 8; ++i) {
          av[i]     = (bf16_t)Ab[klo + i];
          av[8 + i] = (bf16_t)Ab[khi + i];
        }
        if (c == 0) a0 = av; else if (c == 1) a1 = av; else a2 = av;
      }
      const int tj0 = (ti >= 4) ? (ti - 4) : 0;
      for (int tj = tj0; tj <= ti; ++tj) {
        // B tile: B[k][n] = FqR[16*tj+n][k]; per-lane 32B contiguous LDS chunks
        const bf16_t* Bb = fr + (size_t)(tj * 16 + Mrow) * CCH + hi * 16;
        v16bf b0 = *(const v16bf*)(Bb);
        v16bf b1 = *(const v16bf*)(Bb + 32);
        v16bf b2 = *(const v16bf*)(Bb + 64);
        v8f acc = {0.f, 0.f, 0.f, 0.f, 0.f, 0.f, 0.f, 0.f};
        acc = __builtin_amdgcn_wmma_f32_16x16x32_bf16(false, a0, false, b0, (short)0, acc, false, false);
        acc = __builtin_amdgcn_wmma_f32_16x16x32_bf16(false, a1, false, b1, (short)0, acc, false, false);
        acc = __builtin_amdgcn_wmma_f32_16x16x32_bf16(false, a2, false, b2, (short)0, acc, false, false);

        const int koff = ti - tj;                       // diagonal-tile offset 0..4
        const int d0   = koff * 16 + 8 * hi - Mrow;     // dd for r=0
        const int wo0  = ti * 16 + 8 * hi;              // wo for r=0
        float* dst = cvp + d0 * W4 + wo0;               // r stride: 513 floats
        if (koff >= 1 && koff <= 3) {
          // interior diagonals: every (r,lane) element is in-band -> no predication
#pragma unroll
          for (int r = 0; r < 8; ++r) dst[r * (W4 + 1)] = acc[r];
        } else {
          // koff==0 (lower edge) / koff==4 (upper edge): guard band membership
#pragma unroll
          for (int r = 0; r < 8; ++r) {
            const int dd = d0 + r;
            if (dd >= 0 && dd < DP1) dst[r * (W4 + 1)] = acc[r];
          }
        }
      }
    }
  }
  __syncthreads();

  // ---------------- Phase 3: ent_before -> invalid mask ------------------------
  for (int x = tid; x < W4; x += TPB) {
    const int dmax = (x < 63) ? x : 63;
    const int Deff = dmax + 1;
    float entn = 0.0f;
    if (Deff > 1) {
      float m = -1e30f;
      for (int d = 0; d <= dmax; ++d) m = fmaxf(m, cvp[d * W4 + x]);
      float S = 0.0f, E = 0.0f;
      for (int d = 0; d <= dmax; ++d) {
        const float z = (cvp[d * W4 + x] - m) * 10.0f;   // 1/ENT_T
        const float e = __expf(z);
        S += e; E += e * z;
      }
      float ent = __logf(S) - E / S + (float)(DP1 - Deff) * ENT_CLIP_EXTRA;
      entn = ent / (__logf((float)Deff) + 1e-8f);
      entn = fminf(fmaxf(entn, 0.0f), 1.0f);
    }
    inval[x] = (entn > 0.9f) ? 1 : 0;
  }
  __syncthreads();

  // ---------------- Phase 4: in-place prefix sums of cv along w ----------------
  {
    const int d  = tid & 63;
    const int sg = tid >> 6;            // 4 segments of 128
    const int w0 = sg * 128, w1 = w0 + 128;
    float sum = 0.0f;
    for (int w = w0; w < w1; ++w) {
      if (w < d) cvp[d * W4 + w] = 0.0f;  // invalid entries contribute 0
      else       sum += cvp[d * W4 + w];
    }
    red[d * 4 + sg] = sum;
    __syncthreads();
    float run = 0.0f;
    for (int s = 0; s < sg; ++s) run += red[d * 4 + s];
    for (int w = w0; w < w1; ++w) {
      if (w >= d) { run += cvp[d * W4 + w]; cvp[d * W4 + w] = run; }
    }
  }
  __syncthreads();

  // ---------------- Phase 5: refine pixels -> kl / reg / weight ----------------
  float s1t = 0.0f, s2t = 0.0f, zt = 0.0f;
  for (int x = tid; x < W4; x += TPB) {
    if (!inval[x]) continue;            // w==0 outside refine_mask
    // run extents clamped by MAX_HALF=32
    int a_ = (x - 32 > 0) ? x - 32 : 0;
    for (int p = x - 1; p >= x - 33 && p >= 0; --p) {
      if (!inval[p]) { if (p + 1 > a_) a_ = p + 1; break; }
    }
    int b_ = (x + 32 < W4 - 1) ? x + 32 : W4 - 1;
    for (int q = x + 1; q <= x + 33 && q <= W4 - 1; ++q) {
      if (!inval[q]) { if (q - 1 < b_) b_ = q - 1; break; }
    }
    const int DeffA = (b_ + 1 < DP1) ? b_ + 1 : DP1;

    // pass A: agg max + top-2 (agg_d = windowed mean over finite entries)
    float mt = -1e30f, v1 = -1e30f, v2 = -1e30f;
    int i1 = 0, i2 = 0;
    for (int d = 0; d < DP1; ++d) {
      float aggd = -1e30f;
      if (d <= b_) {
        const int   lo  = (a_ > d) ? a_ : d;
        const float sub = (a_ > 0) ? cvp[d * W4 + (a_ - 1)] : 0.0f;
        aggd = (cvp[d * W4 + b_] - sub) / (float)(b_ - lo + 1);
        mt = fmaxf(mt, aggd);
      }
      if (aggd > v1)      { v2 = v1; i2 = i1; v1 = aggd; i1 = d; }
      else if (aggd > v2) { v2 = aggd; i2 = d; }
    }
    // pass B: student max
    const float* sp = Stud + (size_t)bb * DP1 * H4 * W4 + (size_t)hh * W4 + x;
    float ms = -1e30f;
    for (int d = 0; d < DP1; ++d) ms = fmaxf(ms, sp[(size_t)d * (H4 * W4)]);
    // pass C: fused teacher/student accumulations
    float S = 0.0f, E1 = 0.0f, Pd = 0.0f, Es = 0.0f, Ss = 0.0f, Ds = 0.0f;
    for (int d = 0; d < DP1; ++d) {
      const float s  = sp[(size_t)d * (H4 * W4)];
      const float es = __expf(s - ms);
      Ss += es; Ds += es * (float)d;
      if (d <= b_) {
        const int   lo  = (a_ > d) ? a_ : d;
        const float sub = (a_ > 0) ? cvp[d * W4 + (a_ - 1)] : 0.0f;
        const float aggd = (cvp[d * W4 + b_] - sub) / (float)(b_ - lo + 1);
        const float z = (aggd - mt) * 10.0f;   // 1/T_TEACH
        const float e = __expf(z);
        S += e; E1 += e * z; Pd += e * (float)d; Es += e * s;
      }
    }
    const float logS  = __logf(S);
    const float logSs = __logf(Ss);
    float entA = (logS - E1 / S) + (float)(DP1 - DeffA) * ENT_CLIP_EXTRA;
    entA = (DeffA > 1) ? entA / (__logf((float)DeffA) + 1e-8f) : 0.0f;
    entA = fminf(fmaxf(entA, 0.0f), 1.0f);

    const float kl     = (E1 / S - logS) - (Es / S - ms - logSs);
    const float disp_t = Pd / S;
    const float disp_s = Ds / Ss;
    const float diff   = disp_s - disp_t;
    const float reg    = sqrtf(diff * diff + 1e-6f);
    const float gap    = (DeffA >= 2) ? fabsf((float)(i1 - i2)) : 0.0f;
    const float gapn   = fminf(gap * 0.25f, 1.0f);          // /GAP_NORM
    const float wl     = fminf((float)(b_ - a_ + 1) * 0.125f, 1.0f);  // /L0
    const float w      = (gapn + (1.0f - entA)) * sqrtf(wl);          // ^W_GAMMA

    s1t += w * kl; s2t += w * reg; zt += w;
  }

  // ---------------- Phase 6: block reduce + device atomics ---------------------
  __syncthreads();
  red[tid] = s1t; red[TPB + tid] = s2t; red[2 * TPB + tid] = zt;
  __syncthreads();
  if (tid == 0) {
    float a = 0.f, b = 0.f, c = 0.f;
    for (int i = 0; i < TPB; ++i) { a += red[i]; b += red[TPB + i]; c += red[2 * TPB + i]; }
    atomicAdd(&ws[0], a);
    atomicAdd(&ws[1], b);
    atomicAdd(&ws[2], c);
  }
}

__global__ void awdl_finalize(const float* ws, float* out) {
  out[0] = (1.0f * ws[0] + 0.5f * ws[1]) / (ws[2] + 1e-8f);  // L_KL, L_REG, EPS
}

extern "C" void kernel_launch(void* const* d_in, const int* in_sizes, int n_in,
                              void* d_out, int out_size, void* d_ws, size_t ws_size,
                              hipStream_t stream) {
  const float* FqL  = (const float*)d_in[0];
  const float* FqR  = (const float*)d_in[1];
  const float* Stud = (const float*)d_in[2];
  float* out = (float*)d_out;
  float* ws  = (float*)d_ws;
  awdl_init<<<1, 1, 0, stream>>>(ws);
  awdl_main<<<dim3(BATCH * NROW), dim3(TPB), SMEM_BYTES, stream>>>(FqL, FqR, Stud, ws);
  awdl_finalize<<<1, 1, 0, stream>>>(ws, out);
}
